// ParallelMHADejavu_46059229283015
// MI455X (gfx1250) — compile-verified, matrix operand order checked by reading, not run on hardware
//
#include <hip/hip_runtime.h>
#include <hip/hip_bf16.h>
#include <stdint.h>
#include <stddef.h>

typedef __bf16 bf16;
typedef __attribute__((ext_vector_type(16))) __bf16 v16bf;
typedef __attribute__((ext_vector_type(8)))  __bf16 v8bf;
typedef __attribute__((ext_vector_type(8)))  float   v8f;

namespace {
constexpr int kB = 2;
constexpr int kS = 2048;
constexpr int kE = 2048;
constexpr int kH = 16;
constexpr int kD = 128;
constexpr int kM = kB * kS;                         // 4096 tokens
constexpr float kScale = 0.08838834764831845f;      // 1/sqrt(128)
}

union F8 { v8f v; float f[8]; };

#if defined(__AMDGCN__) && __has_builtin(__builtin_amdgcn_global_load_async_to_lds_b128)
#define HAVE_ASYNC_LDS 1
typedef int v4i __attribute__((vector_size(16)));
typedef __attribute__((address_space(1))) void gas_void;
typedef __attribute__((address_space(3))) void las_void;
typedef __attribute__((address_space(1))) v4i  gas_v4i;
typedef __attribute__((address_space(3))) v4i  las_v4i;
#else
#define HAVE_ASYNC_LDS 0
#endif

__device__ __forceinline__ v8bf ld8(const bf16* p) { return *(const v8bf*)p; }

// Build a 16-element bf16 WMMA fragment from two contiguous 16-byte loads.
// Caller passes the lane-adjusted base (already includes the 0/8 k-select),
// elements 0..7 <- K ksel..ksel+7, elements 8..15 <- K ksel+16..ksel+23,
// matching the CDNA5 16-bit A/B fragment layout (ISA 7.12.2).
__device__ __forceinline__ v16bf frag16(const bf16* p) {
  v8bf lo = ld8(p);
  v8bf hi = ld8(p + 16);
  return __builtin_shufflevector(lo, hi, 0,1,2,3,4,5,6,7,8,9,10,11,12,13,14,15);
}

__device__ __forceinline__ v8f wmma_bf16(v16bf a, v16bf b, v8f c) {
  // (neg_a, A, neg_b, B, c_mod, C, reuse_a, reuse_b)
  return __builtin_amdgcn_wmma_f32_16x16x32_bf16(false, a, false, b, (short)0, c, false, false);
}

// 16-byte global -> LDS copy; async on CDNA5 (GLOBAL_LOAD_ASYNC_TO_LDS_B128,
// tracked by ASYNCcnt, no VGPR round trip), synchronous fallback otherwise.
__device__ __forceinline__ void cp16_g2l(const bf16* g, bf16* l) {
#if HAVE_ASYNC_LDS
  __builtin_amdgcn_global_load_async_to_lds_b128(
      (gas_v4i*)((gas_void*)(void*)g),
      (las_v4i*)((las_void*)(void*)l), 0, 0);
#else
  *(v8bf*)l = *(const v8bf*)g;
#endif
}

__device__ __forceinline__ void cp_async_wait() {
#if HAVE_ASYNC_LDS
#if __has_builtin(__builtin_amdgcn_s_wait_asynccnt)
  __builtin_amdgcn_s_wait_asynccnt(0);
#else
  asm volatile("s_wait_asynccnt 0" ::: "memory");
#endif
#endif
}

// ---------------------------------------------------------------------------
// f32 -> bf16 elementwise conversion
// ---------------------------------------------------------------------------
__global__ void cvt_f32_bf16(const float* __restrict__ in, bf16* __restrict__ out, size_t n) {
  size_t i = (size_t)blockIdx.x * blockDim.x + threadIdx.x;
  size_t stride = (size_t)gridDim.x * blockDim.x;
  for (; i < n; i += stride) out[i] = (bf16)in[i];
}

// ---------------------------------------------------------------------------
// bf16 GEMM:  C[M,N] = A[M,K] * W[N,K]^T + bias[N]
// 256 threads = 8 waves; block tile 64x128; each wave a 32x32 tile (2x2 WMMA).
// MODE 0: write f32 row-major to outF.
// MODE 1: QKV epilogue -> scatter bf16 Q[b,h,s,d], K[b,h,s,d]*scale, V^T[b,h,d,s].
// ---------------------------------------------------------------------------
template <int MODE>
__global__ __launch_bounds__(256) void gemm_bf16_kernel(
    const bf16* __restrict__ A, const bf16* __restrict__ W,
    const float* __restrict__ bias,
    float* __restrict__ outF,
    bf16* __restrict__ qO, bf16* __restrict__ kO, bf16* __restrict__ vtO,
    int M, int N, int K)
{
  const int lane = threadIdx.x & 31;
  const int wave = threadIdx.x >> 5;
  const int lh   = lane & 15;            // fragment row/col within 16
  const int hi8  = (lane >> 4) << 3;     // 0 or 8: k-select for A/B, row-base for C/D

  const int m0 = blockIdx.x * 64  + (wave >> 2) * 32;
  const int n0 = blockIdx.y * 128 + (wave & 3) * 32;

  F8 acc[2][2];
#pragma unroll
  for (int i = 0; i < 2; ++i)
#pragma unroll
    for (int j = 0; j < 2; ++j)
#pragma unroll
      for (int e = 0; e < 8; ++e) acc[i][j].f[e] = 0.f;

  const bf16* pa0 = A + (size_t)(m0 + lh) * K + hi8;
  const bf16* pa1 = pa0 + (size_t)16 * K;
  const bf16* pb0 = W + (size_t)(n0 + lh) * K + hi8;
  const bf16* pb1 = pb0 + (size_t)16 * K;

  for (int k0 = 0; k0 < K; k0 += 32) {
    __builtin_prefetch(pa0 + k0 + 256, 0, 1);
    __builtin_prefetch(pb0 + k0 + 256, 0, 1);
    v16bf a0 = frag16(pa0 + k0);
    v16bf a1 = frag16(pa1 + k0);
    v16bf b0 = frag16(pb0 + k0);
    v16bf b1 = frag16(pb1 + k0);
    acc[0][0].v = wmma_bf16(a0, b0, acc[0][0].v);
    acc[0][1].v = wmma_bf16(a0, b1, acc[0][1].v);
    acc[1][0].v = wmma_bf16(a1, b0, acc[1][0].v);
    acc[1][1].v = wmma_bf16(a1, b1, acc[1][1].v);
  }

#pragma unroll
  for (int i = 0; i < 2; ++i) {
#pragma unroll
    for (int j = 0; j < 2; ++j) {
      const int col = n0 + j * 16 + lh;
      const float bv = bias[col];
#pragma unroll
      for (int r = 0; r < 8; ++r) {
        const int row = m0 + i * 16 + hi8 + r;     // C layout: lanes<16 rows 0-7, lanes>=16 rows 8-15
        const float val = acc[i][j].f[r] + bv;
        if (MODE == 0) {
          outF[(size_t)row * N + col] = val;
        } else {
          const int b    = row >> 11;              // / kS
          const int s    = row & (kS - 1);
          const int part = col >> 11;              // / kE : 0=Q 1=K 2=V
          const int nin  = col & (kE - 1);
          const int h    = nin >> 7;               // / kD
          const int d    = nin & (kD - 1);
          const size_t bh = (size_t)(b * kH + h);
          if (part == 0) {
            qO[(bh * kS + s) * kD + d] = (bf16)val;
          } else if (part == 1) {
            kO[(bh * kS + s) * kD + d] = (bf16)(val * kScale);
          } else {
            vtO[(bh * kD + d) * kS + s] = (bf16)val; // V transposed: contiguous keys
          }
        }
      }
    }
  }
}

// ---------------------------------------------------------------------------
// Causal flash attention. One block per (b, h, 64-query group); 4 waves each
// own a 16-query tile. K/V 32-key tiles are staged cooperatively into LDS with
// async global->LDS copies and shared by all 4 waves (4x traffic reduction).
// Q/K: [bh][s][d] bf16 (K pre-scaled), V^T: [bh][d][s] bf16.
// Output: bf16 [b][s][h*D+d] for the projection GEMM.
// ---------------------------------------------------------------------------
__global__ __launch_bounds__(128) void attn_kernel(
    const bf16* __restrict__ Q, const bf16* __restrict__ Kmat,
    const bf16* __restrict__ Vt, bf16* __restrict__ O)
{
  constexpr int LDK = 136;  // K-tile row stride in elems (32 rows x 128 d), padded
  constexpr int LDV = 40;   // V-tile row stride in elems (128 rows x 32 keys), padded
  __shared__ __align__(16) bf16 ktile[32 * LDK];   // [key][d]
  __shared__ __align__(16) bf16 vtile[128 * LDV];  // [d][key]
  __shared__ __align__(16) bf16 plds[4][16 * 32];  // per-wave P-tile staging

  const int tid  = threadIdx.x;
  const int lane = tid & 31;
  const int wave = tid >> 5;
  const int lh   = lane & 15;
  const int hi8  = (lane >> 4) << 3;

  const int tgrp = blockIdx.x & (kS / 64 - 1);   // 64-query group within sequence
  const int bh   = blockIdx.x >> 5;              // b*H + h
  const int q0b  = tgrp * 64;
  const int q0   = q0b + wave * 16;

  const bf16* qp = Q    + (size_t)bh * kS * kD;
  const bf16* kp = Kmat + (size_t)bh * kS * kD;
  const bf16* vp = Vt   + (size_t)bh * kD * kS;

  v16bf qf[4];
#pragma unroll
  for (int c = 0; c < 4; ++c)
    qf[c] = frag16(qp + (size_t)(q0 + lh) * kD + c * 32 + hi8);

  F8 of[8];
  float mrow[8], lrow[8];
#pragma unroll
  for (int c = 0; c < 8; ++c)
#pragma unroll
    for (int e = 0; e < 8; ++e) of[c].f[e] = 0.f;
#pragma unroll
  for (int r = 0; r < 8; ++r) { mrow[r] = -3.0e38f; lrow[r] = 0.f; }

  bf16* myp = &plds[wave][0];

  const int jend = q0 + 16;   // this wave's causal key bound (exclusive)
  const int jmax = q0b + 64;  // block-wide bound: all waves run the full loop
  for (int j0 = 0; j0 < jmax; j0 += 32) {
    // --- cooperative async staging of K (32x128) and V^T (128x32) tiles ---
#pragma unroll
    for (int i = 0; i < 4; ++i) {
      const int s   = tid + 128 * i;         // 0..511 16-byte segments
      const int krw = s >> 4, dsg = s & 15;  // K: 32 rows x 16 segs
      cp16_g2l(kp + (size_t)(j0 + krw) * kD + dsg * 8, &ktile[krw * LDK + dsg * 8]);
      const int vrw = s >> 2, ksg = s & 3;   // V: 128 rows x 4 segs
      cp16_g2l(vp + (size_t)vrw * kS + j0 + ksg * 8, &vtile[vrw * LDV + ksg * 8]);
    }
    cp_async_wait();
    __syncthreads();

    if (j0 < jend) {   // wave-uniform: waves past their causal range skip compute
      // --- scores S = Q K^T for 16 rows x 32 keys (two 16-col halves) ---
      F8 s[2];
#pragma unroll
      for (int half = 0; half < 2; ++half) {
#pragma unroll
        for (int e = 0; e < 8; ++e) s[half].f[e] = 0.f;
#pragma unroll
        for (int c = 0; c < 4; ++c) {
          v16bf kf = frag16(&ktile[(half * 16 + lh) * LDK + c * 32 + hi8]);
          s[half].v = wmma_bf16(qf[c], kf, s[half].v);
        }
      }
      // --- causal mask (reference-style additive -10000) ---
#pragma unroll
      for (int half = 0; half < 2; ++half) {
        const int j = j0 + half * 16 + lh;
#pragma unroll
        for (int r = 0; r < 8; ++r) {
          const int i = q0 + hi8 + r;
          if (j > i) s[half].f[r] -= 10000.0f;
        }
      }
      // --- online softmax; each row lives in one VGPR of one 16-lane half ---
#pragma unroll
      for (int r = 0; r < 8; ++r) {
        float tmax = fmaxf(s[0].f[r], s[1].f[r]);
#pragma unroll
        for (int mm = 1; mm < 16; mm <<= 1)
          tmax = fmaxf(tmax, __shfl_xor(tmax, mm, 32));
        const float mnew = fmaxf(mrow[r], tmax);
        const float corr = __expf(mrow[r] - mnew);
        const float p0 = __expf(s[0].f[r] - mnew);
        const float p1 = __expf(s[1].f[r] - mnew);
        float psum = p0 + p1;
#pragma unroll
        for (int mm = 1; mm < 16; mm <<= 1)
          psum += __shfl_xor(psum, mm, 32);
        lrow[r] = lrow[r] * corr + psum;
        mrow[r] = mnew;
#pragma unroll
        for (int c = 0; c < 8; ++c) of[c].f[r] *= corr;
        // stage P (bf16) to this wave's LDS tile, row-major [row][key]
        myp[(hi8 + r) * 32 + lh]      = (bf16)p0;
        myp[(hi8 + r) * 32 + 16 + lh] = (bf16)p1;
      }
      asm volatile("s_wait_dscnt 0" ::: "memory");  // per-wave fence for P staging
      const v16bf pf = frag16(myp + lh * 32 + hi8); // re-read P in A-fragment layout
      // --- O += P V over this key block; V^T tile gives contiguous B frags ---
#pragma unroll
      for (int c = 0; c < 8; ++c) {
        v16bf vf = frag16(&vtile[(c * 16 + lh) * LDV + hi8]);
        of[c].v = wmma_bf16(pf, vf, of[c].v);
      }
    }
    __syncthreads();   // protect K/V tile WAR before next staging round
  }

  const int b = bh >> 4;
  const int h = bh & (kH - 1);
#pragma unroll
  for (int c = 0; c < 8; ++c) {
    const int d = c * 16 + lh;
#pragma unroll
    for (int r = 0; r < 8; ++r) {
      const int row = q0 + hi8 + r;
      const float val = of[c].f[r] / lrow[r];
      O[((size_t)(b * kS + row)) * kE + h * kD + d] = (bf16)val;
    }
  }
}

// ---------------------------------------------------------------------------
extern "C" void kernel_launch(void* const* d_in, const int* in_sizes, int n_in,
                              void* d_out, int out_size, void* d_ws, size_t ws_size,
                              hipStream_t stream) {
  (void)in_sizes; (void)n_in; (void)out_size; (void)ws_size;

  const float* x    = (const float*)d_in[0];   // [B,S,E]
  const float* wqkv = (const float*)d_in[1];   // [3E,E]
  const float* bqkv = (const float*)d_in[2];   // [3E]
  const float* wout = (const float*)d_in[3];   // [E,E]
  const float* bout = (const float*)d_in[4];   // [E]
  float* out = (float*)d_out;                  // [B,S,E] f32

  // Workspace partition (all bf16): ~112 MiB total
  bf16* p = (bf16*)d_ws;
  bf16* xb    = p; p += (size_t)kM * kE;       // x in bf16
  bf16* wqkvb = p; p += (size_t)3 * kE * kE;   // wqkv in bf16
  bf16* woutb = p; p += (size_t)kE * kE;       // wout in bf16
  bf16* qb    = p; p += (size_t)kM * kE;       // Q [bh][s][d]
  bf16* kb    = p; p += (size_t)kM * kE;       // K [bh][s][d] (pre-scaled)
  bf16* vtb   = p; p += (size_t)kM * kE;       // V^T [bh][d][s]
  bf16* attnb = p; p += (size_t)kM * kE;       // attention output [b][s][e]

  cvt_f32_bf16<<<2048, 256, 0, stream>>>(x, xb, (size_t)kM * kE);
  cvt_f32_bf16<<<2048, 256, 0, stream>>>(wqkv, wqkvb, (size_t)3 * kE * kE);
  cvt_f32_bf16<<<1024, 256, 0, stream>>>(wout, woutb, (size_t)kE * kE);

  // QKV projection: M=4096, N=6144, K=2048 -> scatter Q/K/V^T
  gemm_bf16_kernel<1><<<dim3(kM / 64, (3 * kE) / 128), 256, 0, stream>>>(
      xb, wqkvb, bqkv, nullptr, qb, kb, vtb, kM, 3 * kE, kE);

  // Flash attention: B*H*(S/64) = 1024 blocks of 4 waves
  attn_kernel<<<dim3(kB * kH * (kS / 64)), 128, 0, stream>>>(qb, kb, vtb, attnb);

  // Output projection: M=4096, N=2048, K=2048 -> f32 out + bias
  gemm_bf16_kernel<0><<<dim3(kM / 64, kE / 128), 256, 0, stream>>>(
      attnb, woutb, bout, out, nullptr, nullptr, nullptr, kM, kE, kE);
}
